// FeatureNetLoss_45913200394829
// MI455X (gfx1250) — matrix operand order repaired and checked by reading, not compiled, not run on hardware
//
#include <hip/hip_runtime.h>
#include <hip/hip_bf16.h>

constexpr int Bc = 8, Nc = 500, Dc = 256, Hc = 480, Wc = 640;
constexpr int NPAD = 512;          // 500 padded to multiple of 32 (zero rows)
constexpr int CM = 8192;           // candidate capacity per batch (pow2 for bitonic)

typedef __attribute__((ext_vector_type(8)))  _Float16 v8h;
typedef __attribute__((ext_vector_type(16))) _Float16 v16h;
typedef __attribute__((ext_vector_type(8)))  float    v8f;

typedef __attribute__((ext_vector_type(4))) unsigned int u32x4;
typedef __attribute__((ext_vector_type(8))) int          i32x8;
typedef __attribute__((ext_vector_type(4))) int          i32x4;

#if __has_builtin(__builtin_amdgcn_tensor_load_to_lds)
#define HAVE_TDM 1
#else
#define HAVE_TDM 0
#endif

// ---------------- WMMA helpers (CDNA5 wave32, 16x16x32 f16) ----------------

__device__ __forceinline__ v8f wmma16(v16h a, v16h b, v8f c) {
  // 8 args: (neg_a, A, neg_b, B, c_mod, C, reuse_a, reuse_b)
  return __builtin_amdgcn_wmma_f32_16x16x32_f16(false, a, false, b, (short)0, c,
                                                false, false);
}

// A operand: 16x32 (MxK). Lanes 0-15 -> M=lane, K 0..15 contiguous; lanes 16-31
// -> K 16..31.
__device__ __forceinline__ v16h fragA(const _Float16* __restrict__ base, int n0,
                                      int k0, int lane) {
  int row = lane & 15;
  int kh  = lane >> 4;  // 0 or 1
  const _Float16* rp = base + (size_t)(n0 + row) * Dc + k0 + kh * 16;
  union { v16h v; v8h h[2]; } u;
  u.h[0] = *(const v8h*)(rp);
  u.h[1] = *(const v8h*)(rp + 8);
  return u.v;
}

// B operand: 32x16 (KxN). Lanes 0-15 -> N=lane, K 0..15 contiguous; lanes 16-31
// -> K 16..31.
__device__ __forceinline__ v16h fragB(const _Float16* __restrict__ base, int m0,
                                      int k0, int lane) {
  int col = lane & 15;
  int kb  = (lane >> 4) * 16;
  const _Float16* rp = base + (size_t)(m0 + col) * Dc + k0 + kb;
  union { v16h v; v8h h[2]; } u;
  u.h[0] = *(const v8h*)(rp);
  u.h[1] = *(const v8h*)(rp + 8);
  return u.v;
}

__device__ __forceinline__ float wave_sum(float v) {
#pragma unroll
  for (int off = 16; off > 0; off >>= 1) v += __shfl_down(v, off, 32);
  return v;
}

// ---------------- Tensor Data Mover: 2-D tile (rows x Dc f16) -> LDS ----------------

#if HAVE_TDM
__device__ __forceinline__ void tdm_load_2d(void* lds_ptr, const void* gptr, int rows) {
  unsigned long long ga = (unsigned long long)(size_t)gptr;
  unsigned lds_off = (unsigned)(size_t)lds_ptr;  // flat shared addr low bits = LDS offset
  u32x4 g0;
  g0[0] = 1u;                                              // count=1, user descriptor
  g0[1] = lds_off;                                         // lds_addr
  g0[2] = (unsigned)(ga & 0xFFFFFFFFull);                  // global_addr[31:0]
  g0[3] = (unsigned)((ga >> 32) & 0x1FFFFFFull) | (2u << 30);  // addr[56:32], type=2
  i32x8 g1;
  g1[0] = (int)(1u << 16);                                 // data_size=1 (2 bytes)
  g1[1] = (int)(((unsigned)Dc & 0xFFFFu) << 16);           // tensor_dim0[15:0] @bit48
  g1[2] = 0;                                               // tensor_dim0 hi, tensor_dim1 lo
  g1[3] = (int)((((unsigned)Dc & 0xFFFFu) << 16) | 0x10u); // tile_dim0=Dc, tensor_dim1=2^20
  g1[4] = rows;                                            // tile_dim1=rows, tile_dim2=0
  g1[5] = Dc;                                              // tensor_dim0_stride[31:0]
  g1[6] = 0;
  g1[7] = 0;
  i32x4 z4 = {0, 0, 0, 0};
#if defined(__clang_major__) && (__clang_major__ >= 23)
  i32x8 z8 = {0, 0, 0, 0, 0, 0, 0, 0};
  __builtin_amdgcn_tensor_load_to_lds(g0, g1, z4, z4, z8, 0);
#else
  __builtin_amdgcn_tensor_load_to_lds(g0, g1, z4, z4, 0);
#endif
}
#endif

// ---------------- init ----------------

__global__ void k_init(unsigned* __restrict__ target,
                       unsigned long long* __restrict__ cand_key,
                       int* __restrict__ cand_cnt, float* __restrict__ acc) {
  int t = blockIdx.x * blockDim.x + threadIdx.x;
  if (t < Bc * Hc * Wc) target[t] = 0u;
  if (t < Bc * CM) cand_key[t] = 0ull;
  if (t < Bc) cand_cnt[t] = 0;
  if (t < 8) acc[t] = 0.f;
}

// ---------------- descriptor normalize + f16 pack (padded) ----------------

__global__ void k_norm(const float* __restrict__ desc, _Float16* __restrict__ dn) {
  int t = blockIdx.x * blockDim.x + threadIdx.x;
  if (t >= Bc * NPAD) return;
  int b = t / NPAD, n = t % NPAD;
  _Float16* out = dn + (size_t)t * Dc;
  if (n >= Nc) {
    for (int k = 0; k < Dc; ++k) out[k] = (_Float16)0.f;
    return;
  }
  const float* src = desc + ((size_t)b * Nc + n) * Dc;
  float ss = 0.f;
  for (int k = 0; k < Dc; ++k) ss += src[k] * src[k];
  float inv = 1.f / fmaxf(sqrtf(ss), 1e-8f);
  for (int k = 0; k < Dc; ++k) out[k] = (_Float16)(src[k] * inv);
}

// ---------------- distinction loss GEMM (per wave: 32x32 macro-tile, global path) ----------------

__device__ __forceinline__ float relu_tile(const v8f& c, int n0, int m0, int lane) {
  int m  = m0 + (lane & 15);
  int nb = n0 + ((lane >> 4) << 3);
  float s = 0.f;
  if (m < Nc) {
#pragma unroll
    for (int r = 0; r < 8; ++r)
      if (nb + r < Nc) s += fmaxf(c[r], 0.f);
  }
  return s;
}

__global__ void k_gemm_dist(const _Float16* __restrict__ dn, float* __restrict__ acc) {
  int wave = (int)((blockIdx.x * blockDim.x + threadIdx.x) >> 5);
  int lane = threadIdx.x & 31;
  const int nt = NPAD / 32, perb = nt * nt;  // 16, 256
  if (wave >= Bc * perb) return;
  int b = wave / perb, t = wave % perb;
  int n0 = (t / nt) * 32, m0 = (t % nt) * 32;
  const _Float16* base = dn + (size_t)b * NPAD * Dc;
  v8f c00 = {}, c01 = {}, c10 = {}, c11 = {};
  for (int k0 = 0; k0 < Dc; k0 += 32) {
    if (k0 + 32 < Dc) {  // gfx1250 global_prefetch_b8 of next K-slice
      __builtin_prefetch((const void*)(base + (size_t)(n0 + (lane & 15)) * Dc + k0 + 32), 0, 1);
      __builtin_prefetch((const void*)(base + (size_t)(m0 + (lane & 15)) * Dc + k0 + 32), 0, 1);
    }
    v16h a0 = fragA(base, n0, k0, lane);
    v16h a1 = fragA(base, n0 + 16, k0, lane);
    v16h b0 = fragB(base, m0, k0, lane);
    v16h b1 = fragB(base, m0 + 16, k0, lane);
    c00 = wmma16(a0, b0, c00);
    c01 = wmma16(a0, b1, c01);
    c10 = wmma16(a1, b0, c10);
    c11 = wmma16(a1, b1, c11);
  }
  float s = relu_tile(c00, n0, m0, lane) + relu_tile(c01, n0, m0 + 16, lane) +
            relu_tile(c10, n0 + 16, m0, lane) + relu_tile(c11, n0 + 16, m0 + 16, lane);
  s = wave_sum(s);
  if (lane == 0 && s != 0.f) atomicAdd(&acc[0], s);
}

// ---------------- pairwise projection (shared by points & corners) ----------------

__device__ __forceinline__ float bilinear(const float* __restrict__ img, float px, float py) {
  float x = fminf(fmaxf(px, 0.f), (float)(Wc - 1));
  float y = fminf(fmaxf(py, 0.f), (float)(Hc - 1));
  float x0 = floorf(x), y0 = floorf(y);
  float x1 = fminf(x0 + 1.f, (float)(Wc - 1));
  float y1 = fminf(y0 + 1.f, (float)(Hc - 1));
  float wx = x - x0, wy = y - y0;
  int x0i = (int)x0, x1i = (int)x1, y0i = (int)y0, y1i = (int)y1;
  float v00 = img[y0i * Wc + x0i], v01 = img[y0i * Wc + x1i];
  float v10 = img[y1i * Wc + x0i], v11 = img[y1i * Wc + x1i];
  return v00 * (1.f - wx) * (1.f - wy) + v01 * wx * (1.f - wy) +
         v10 * (1.f - wx) * wy + v11 * wx * wy;
}

__global__ void k_project(const float* __restrict__ pts, const float* __restrict__ depths,
                          const float* __restrict__ poses, const float* __restrict__ Km,
                          float* __restrict__ proj, int* __restrict__ invis) {
  int t = blockIdx.x * blockDim.x + threadIdx.x;
  if (t >= Bc * Nc) return;
  int i = t / Nc, n = t % Nc;
  float xn = pts[(i * Nc + n) * 2 + 0], yn = pts[(i * Nc + n) * 2 + 1];
  float px = (xn + 1.f) * 0.5f * (Wc - 1);
  float py = (yn + 1.f) * 0.5f * (Hc - 1);
  float d = bilinear(depths + (size_t)i * Hc * Wc, px, py);
  // K^-1 (general 3x3 adjugate)
  const float* KK = Km + i * 9;
  float a = KK[0], bq = KK[1], c = KK[2], dd = KK[3], e = KK[4], f = KK[5];
  float g = KK[6], h = KK[7], ii = KK[8];
  float A0 = e * ii - f * h, B0 = -(dd * ii - f * g), C0 = dd * h - e * g;
  float det = a * A0 + bq * B0 + c * C0;
  float id = 1.f / det;
  float i00 = A0 * id, i01 = -(bq * ii - c * h) * id, i02 = (bq * f - c * e) * id;
  float i10 = B0 * id, i11 = (a * ii - c * g) * id, i12 = -(a * f - c * dd) * id;
  float i20 = C0 * id, i21 = -(a * h - bq * g) * id, i22 = (a * e - bq * dd) * id;
  float Xc0 = d * (i00 * px + i01 * py + i02);
  float Xc1 = d * (i10 * px + i11 * py + i12);
  float Xc2 = d * (i20 * px + i21 * py + i22);
  const float* P = poses + i * 16;  // row-major 4x4
  float Xw0 = P[0] * Xc0 + P[1] * Xc1 + P[2] * Xc2 + P[3];
  float Xw1 = P[4] * Xc0 + P[5] * Xc1 + P[6] * Xc2 + P[7];
  float Xw2 = P[8] * Xc0 + P[9] * Xc1 + P[10] * Xc2 + P[11];
  bool dbad = (d <= 0.f);
  for (int j = 0; j < Bc; ++j) {
    const float* Q = poses + j * 16;
    float dx = Xw0 - Q[3], dy = Xw1 - Q[7], dz = Xw2 - Q[11];
    // Xj = R_j^T * diff
    float X0 = Q[0] * dx + Q[4] * dy + Q[8] * dz;
    float X1 = Q[1] * dx + Q[5] * dy + Q[9] * dz;
    float X2 = Q[2] * dx + Q[6] * dy + Q[10] * dz;
    const float* Kj = Km + j * 9;
    float u = Kj[0] * X0 + Kj[1] * X1 + Kj[2] * X2;
    float v = Kj[3] * X0 + Kj[4] * X1 + Kj[5] * X2;
    float w = Kj[6] * X0 + Kj[7] * X1 + Kj[8] * X2;
    float zs = (fabsf(w) < 1e-6f) ? 1e-6f : w;
    float pnx = 2.f * (u / zs) / (Wc - 1) - 1.f;
    float pny = 2.f * (v / zs) / (Hc - 1) - 1.f;
    int iv = (w <= 1e-6f) || (fabsf(pnx) > 1.f) || (fabsf(pny) > 1.f) || dbad;
    size_t o = ((size_t)i * Bc + j) * Nc + n;
    proj[o * 2 + 0] = pnx;
    proj[o * 2 + 1] = pny;
    invis[o] = iv;
  }
}

// ---------------- match loss GEMM: TDM-staged LDS tiles + fused sparse-mask epilogue ----------------

__device__ __forceinline__ void match_tile(const v8f& c, int a, int b, int n0, int m0,
                                           int lane, const float* __restrict__ points,
                                           const float* __restrict__ proj,
                                           const int* __restrict__ invis,
                                           float& msum, float& mcnt) {
  int m = m0 + (lane & 15);
  if (m >= Nc) return;
  const float* pd = proj + (((size_t)a * Bc + b) * Nc + m) * 2;
  float dxp = (pd[0] + 1.f) * 0.5f * (Wc - 1);
  float dyp = (pd[1] + 1.f) * 0.5f * (Hc - 1);
  int nb = n0 + ((lane >> 4) << 3);
#pragma unroll
  for (int r = 0; r < 8; ++r) {
    int n = nb + r;
    if (n >= Nc || n >= m) continue;
    if (invis[((size_t)a * Bc + b) * Nc + n]) continue;
    const float* ps = points + ((size_t)b * Nc + n) * 2;
    float sx = (ps[0] + 1.f) * 0.5f * (Wc - 1);
    float sy = (ps[1] + 1.f) * 0.5f * (Hc - 1);
    float ex = sx - dxp, ey = sy - dyp;
    if (ex * ex + ey * ey <= 1.0f) {
      msum += 1.f - c[r];
      mcnt += 1.f;
    }
  }
}

// Block = 128 threads (4 waves), one 64x64 region of one (a,b) pair.
// Both operand panels (64 rows x 256 f16 each, 32 KB) staged in LDS by the TDM.
__global__ void __launch_bounds__(128) k_gemm_match(
    const _Float16* __restrict__ dn, const float* __restrict__ points,
    const float* __restrict__ proj, const int* __restrict__ invis,
    float* __restrict__ acc) {
  __shared__ __align__(16) _Float16 As[64 * Dc];  // 32 KB
  __shared__ __align__(16) _Float16 Bs[64 * Dc];  // 32 KB
  int pair = blockIdx.x >> 6;       // 64 regions per pair
  int reg  = blockIdx.x & 63;
  int a = pair >> 3, b = pair & 7;  // Bc = 8
  int nR = (reg >> 3) * 64;         // 8 x 8 regions over 512x512
  int mR = (reg & 7) * 64;
  const _Float16* gA = dn + ((size_t)b * NPAD + nR) * Dc;
  const _Float16* gB = dn + ((size_t)a * NPAD + mR) * Dc;
#if HAVE_TDM
  if (threadIdx.x < 32) {           // wave 0 issues both tensor DMAs
    tdm_load_2d(As, gA, 64);
    tdm_load_2d(Bs, gB, 64);
    __builtin_amdgcn_s_wait_tensorcnt(0);
  }
#else
  for (int i = threadIdx.x * 8; i < 64 * Dc; i += blockDim.x * 8) {
    *(v8h*)(As + i) = *(const v8h*)(gA + i);
    *(v8h*)(Bs + i) = *(const v8h*)(gB + i);
  }
#endif
  __syncthreads();
  int wv   = threadIdx.x >> 5;      // 0..3 -> 2x2 wave grid of 32x32 tiles
  int lane = threadIdx.x & 31;
  int nL = (wv >> 1) * 32;          // local offsets within LDS panels
  int mL = (wv & 1) * 32;
  v8f c00 = {}, c01 = {}, c10 = {}, c11 = {};
  for (int k0 = 0; k0 < Dc; k0 += 32) {
    v16h a0 = fragA(As, nL, k0, lane);
    v16h a1 = fragA(As, nL + 16, k0, lane);
    v16h b0 = fragB(Bs, mL, k0, lane);
    v16h b1 = fragB(Bs, mL + 16, k0, lane);
    c00 = wmma16(a0, b0, c00);
    c01 = wmma16(a0, b1, c01);
    c10 = wmma16(a1, b0, c10);
    c11 = wmma16(a1, b1, c11);
  }
  int n0 = nR + nL, m0 = mR + mL;   // global tile coordinates
  float msum = 0.f, mcnt = 0.f;
  match_tile(c00, a, b, n0, m0, lane, points, proj, invis, msum, mcnt);
  match_tile(c01, a, b, n0, m0 + 16, lane, points, proj, invis, msum, mcnt);
  match_tile(c10, a, b, n0 + 16, m0, lane, points, proj, invis, msum, mcnt);
  match_tile(c11, a, b, n0 + 16, m0 + 16, lane, points, proj, invis, msum, mcnt);
  msum = wave_sum(msum);
  mcnt = wave_sum(mcnt);
  if (lane == 0 && mcnt != 0.f) {
    atomicAdd(&acc[1], msum);
    atomicAdd(&acc[2], mcnt);
  }
}

// ---------------- image pipeline ----------------

__global__ void k_gray(const float* __restrict__ imgs, float* __restrict__ gray) {
  int t = blockIdx.x * blockDim.x + threadIdx.x;
  if (t >= Bc * Hc * Wc) return;
  int b = t / (Hc * Wc), p = t % (Hc * Wc);
  const float* im = imgs + (size_t)b * 3 * Hc * Wc;
  gray[t] = 0.299f * im[p] + 0.587f * im[Hc * Wc + p] + 0.114f * im[2 * Hc * Wc + p];
}

__global__ void k_grad(const float* __restrict__ gray, float* __restrict__ gxo,
                       float* __restrict__ gyo) {
  int t = blockIdx.x * blockDim.x + threadIdx.x;
  if (t >= Bc * Hc * Wc) return;
  int b = t / (Hc * Wc), p = t % (Hc * Wc);
  int y = p / Wc, x = p % Wc;
  const float* G = gray + (size_t)b * Hc * Wc;
  const float s[3][3] = {{-1.f, 0.f, 1.f}, {-2.f, 0.f, 2.f}, {-1.f, 0.f, 1.f}};
  float ax = 0.f, ay = 0.f;
  for (int u = 0; u < 3; ++u) {
    int yy = y + u - 1;
    if (yy < 0 || yy >= Hc) continue;
    for (int v = 0; v < 3; ++v) {
      int xx = x + v - 1;
      if (xx < 0 || xx >= Wc) continue;
      float g = G[yy * Wc + xx];
      ax += s[u][v] * g;
      ay += s[v][u] * g;
    }
  }
  gxo[t] = ax * 0.125f;
  gyo[t] = ay * 0.125f;
}

__global__ void k_resp(const float* __restrict__ gxb, const float* __restrict__ gyb,
                       float* __restrict__ resp) {
  int t = blockIdx.x * blockDim.x + threadIdx.x;
  if (t >= Bc * Hc * Wc) return;
  int b = t / (Hc * Wc), p = t % (Hc * Wc);
  int y = p / Wc, x = p % Wc;
  const float g1[7] = {0.00443305f, 0.05400558f, 0.24203623f, 0.39905030f,
                       0.24203623f, 0.05400558f, 0.00443305f};
  const float* GX = gxb + (size_t)b * Hc * Wc;
  const float* GY = gyb + (size_t)b * Hc * Wc;
  float dx2 = 0.f, dy2 = 0.f, dxy = 0.f;
  for (int u = 0; u < 7; ++u) {
    int yy = y + u - 3;
    if (yy < 0 || yy >= Hc) continue;
    for (int v = 0; v < 7; ++v) {
      int xx = x + v - 3;
      if (xx < 0 || xx >= Wc) continue;
      float w = g1[u] * g1[v];
      float gx_ = GX[yy * Wc + xx], gy_ = GY[yy * Wc + xx];
      dx2 += w * gx_ * gx_;
      dy2 += w * gy_ * gy_;
      dxy += w * gx_ * gy_;
    }
  }
  float tr = dx2 + dy2;
  float det = dx2 * dy2 - dxy * dxy;
  float disc = tr * tr - 4.f * det;
  resp[t] = 0.5f * (tr - sqrtf(fmaxf(disc, 0.f)));
}

__global__ void k_nms(const float* __restrict__ resp, float* __restrict__ nmsv) {
  int t = blockIdx.x * blockDim.x + threadIdx.x;
  if (t >= Bc * Hc * Wc) return;
  int b = t / (Hc * Wc), p = t % (Hc * Wc);
  int y = p / Wc, x = p % Wc;
  const float* R = resp + (size_t)b * Hc * Wc;
  float v = R[y * Wc + x];
  float mp = -3.402823e38f;
  for (int dy = -2; dy <= 2; ++dy) {
    int yy = y + dy;
    if (yy < 0 || yy >= Hc) continue;
    for (int dx = -2; dx <= 2; ++dx) {
      int xx = x + dx;
      if (xx < 0 || xx >= Wc) continue;
      mp = fmaxf(mp, R[yy * Wc + xx]);
    }
  }
  nmsv[t] = (v == mp) ? v : 0.f;
}

// one 8x8 block per workgroup: keep pixels equal to block max, append as sort keys
__global__ void k_blockcand(const float* __restrict__ nmsv,
                            unsigned long long* __restrict__ cand_key,
                            int* __restrict__ cand_cnt) {
  int b = blockIdx.z;
  int y = blockIdx.y * 8 + (threadIdx.x >> 3);
  int x = blockIdx.x * 8 + (threadIdx.x & 7);
  float v = nmsv[(size_t)b * Hc * Wc + y * Wc + x];
  __shared__ float red[64];
  red[threadIdx.x] = v;
  __syncthreads();
  for (int s = 32; s > 0; s >>= 1) {
    if ((int)threadIdx.x < s) red[threadIdx.x] = fmaxf(red[threadIdx.x], red[threadIdx.x + s]);
    __syncthreads();
  }
  float bm = red[0];
  if (v > 0.f && v == bm) {
    int c = atomicAdd(&cand_cnt[b], 1);
    if (c < CM) {
      int idx = y * Wc + x;
      cand_key[(size_t)b * CM + c] =
          ((unsigned long long)__float_as_uint(v) << 32) |
          (unsigned long long)(0xFFFFFFFFu - (unsigned)idx);
    }
  }
}

// global-memory bitonic sort (descending), one workgroup per batch; emit top 500
__global__ void k_topk(unsigned long long* __restrict__ cand_key,
                       float* __restrict__ cvals, float* __restrict__ cpts) {
  int b = blockIdx.x;
  unsigned long long* keys = cand_key + (size_t)b * CM;
  unsigned tid = threadIdx.x, bs = blockDim.x;
  for (unsigned k = 2; k <= (unsigned)CM; k <<= 1) {
    for (unsigned j = k >> 1; j > 0; j >>= 1) {
      __syncthreads();
      for (unsigned i = tid; i < (unsigned)CM; i += bs) {
        unsigned l = i ^ j;
        if (l > i) {
          unsigned long long a = keys[i], c = keys[l];
          bool up = ((i & k) == 0);
          bool sw = up ? (a < c) : (a > c);
          if (sw) { keys[i] = c; keys[l] = a; }
        }
      }
    }
  }
  __syncthreads();
  for (unsigned i = tid; i < (unsigned)Nc; i += bs) {
    unsigned long long kk = keys[i];
    float v;
    int id;
    if (kk == 0ull) { v = 0.f; id = 0; }
    else {
      v = __uint_as_float((unsigned)(kk >> 32));
      id = (int)(0xFFFFFFFFu - (unsigned)(kk & 0xFFFFFFFFull));
    }
    cvals[b * Nc + i] = v;
    int x = id % Wc, y = id / Wc;
    cpts[(b * Nc + i) * 2 + 0] = 2.f * (float)x / (Wc - 1) - 1.f;
    cpts[(b * Nc + i) * 2 + 1] = 2.f * (float)y / (Hc - 1) - 1.f;
  }
}

// scatter-max corner values into target images (float bits, all values >= 0)
__global__ void k_scatter(const float* __restrict__ projc, const int* __restrict__ invisc,
                          const float* __restrict__ cvals, unsigned* __restrict__ target) {
  int t = blockIdx.x * blockDim.x + threadIdx.x;
  if (t >= Bc * Bc * Nc) return;
  int i = t / (Bc * Nc);
  int rem = t % (Bc * Nc);
  int j = rem / Nc, n = rem % Nc;
  size_t o = ((size_t)i * Bc + j) * Nc + n;
  if (invisc[o]) return;
  float px = (projc[o * 2 + 0] + 1.f) * 0.5f * (Wc - 1);
  float py = (projc[o * 2 + 1] + 1.f) * 0.5f * (Hc - 1);
  int wq = (int)rintf(px), hq = (int)rintf(py);
  if (wq < 0) return;
  int hc = min(max(hq, 0), Hc - 1);
  int wc = min(max(wq, 0), Wc - 1);
  float v = cvals[i * Nc + n];
  if (v > 0.f)
    atomicMax(&target[(size_t)j * Hc * Wc + hc * Wc + wc], __float_as_uint(v));
}

// final 5x5 NMS on target + BCE vs pointness (fused)
__global__ void k_bce(const unsigned* __restrict__ target,
                      const float* __restrict__ pointness, float* __restrict__ acc) {
  int t = blockIdx.x * blockDim.x + threadIdx.x;
  float local = 0.f;
  if (t < Bc * Hc * Wc) {
    int b = t / (Hc * Wc), p = t % (Hc * Wc);
    int y = p / Wc, x = p % Wc;
    const unsigned* T = target + (size_t)b * Hc * Wc;
    unsigned vb = T[y * Wc + x];
    unsigned mp = 0u;
    for (int dy = -2; dy <= 2; ++dy) {
      int yy = y + dy;
      if (yy < 0 || yy >= Hc) continue;
      for (int dx = -2; dx <= 2; ++dx) {
        int xx = x + dx;
        if (xx < 0 || xx >= Wc) continue;
        unsigned q = T[yy * Wc + xx];
        mp = (q > mp) ? q : mp;  // bit order == float order for non-negative
      }
    }
    float tv = __uint_as_float(vb);
    int tt = (tv > 0.f && vb == mp) ? 1 : 0;
    float pv = pointness[t];
    float lp = fmaxf(logf(pv), -100.f);
    float l1 = fmaxf(logf(1.f - pv), -100.f);
    local = -(tt ? lp : l1);
  }
  local = wave_sum(local);
  if ((threadIdx.x & 31) == 0 && local != 0.f) atomicAdd(&acc[3], local);
}

__global__ void k_final(const float* __restrict__ acc, float* __restrict__ out) {
  out[0] = acc[0] / ((float)Bc * Nc * Nc)        // distinction mean
         + acc[3] / ((float)Bc * Hc * Wc)        // BCE mean
         + acc[1] / fmaxf(acc[2], 1.f);          // match loss
}

// ---------------- launch ----------------

extern "C" void kernel_launch(void* const* d_in, const int* in_sizes, int n_in,
                              void* d_out, int out_size, void* d_ws, size_t ws_size,
                              hipStream_t stream) {
  (void)in_sizes; (void)n_in; (void)out_size; (void)ws_size;
  const float* desc      = (const float*)d_in[0];
  const float* points    = (const float*)d_in[1];
  const float* pointness = (const float*)d_in[2];
  const float* depths    = (const float*)d_in[3];
  const float* poses     = (const float*)d_in[4];
  const float* Km        = (const float*)d_in[5];
  const float* imgs      = (const float*)d_in[6];
  float* out = (float*)d_out;

  char* w = (char*)d_ws;
  size_t off = 0;
  auto take = [&](size_t bytes) -> char* {
    char* p = w + off;
    off = (off + bytes + 255) & ~(size_t)255;
    return p;
  };
  _Float16* dn      = (_Float16*)take((size_t)Bc * NPAD * Dc * sizeof(_Float16));
  float* proj_p     = (float*)take((size_t)Bc * Bc * Nc * 2 * sizeof(float));
  int*   invis_p    = (int*)take((size_t)Bc * Bc * Nc * sizeof(int));
  float* proj_c     = (float*)take((size_t)Bc * Bc * Nc * 2 * sizeof(float));
  int*   invis_c    = (int*)take((size_t)Bc * Bc * Nc * sizeof(int));
  float* cpts       = (float*)take((size_t)Bc * Nc * 2 * sizeof(float));
  float* cvals      = (float*)take((size_t)Bc * Nc * sizeof(float));
  unsigned long long* cand_key = (unsigned long long*)take((size_t)Bc * CM * 8);
  int*   cand_cnt   = (int*)take(Bc * sizeof(int));
  float* acc        = (float*)take(8 * sizeof(float));
  float* gray       = (float*)take((size_t)Bc * Hc * Wc * sizeof(float));
  float* gxb        = (float*)take((size_t)Bc * Hc * Wc * sizeof(float));
  float* gyb        = (float*)take((size_t)Bc * Hc * Wc * sizeof(float));
  float* resp       = (float*)take((size_t)Bc * Hc * Wc * sizeof(float));
  float* nmsv       = (float*)take((size_t)Bc * Hc * Wc * sizeof(float));
  unsigned* target  = (unsigned*)take((size_t)Bc * Hc * Wc * sizeof(unsigned));

  const int HWB = Bc * Hc * Wc;
  k_init<<<(HWB + 255) / 256, 256, 0, stream>>>(target, cand_key, cand_cnt, acc);
  k_norm<<<(Bc * NPAD + 255) / 256, 256, 0, stream>>>(desc, dn);
  // distinction: Bc*256 waves (32x32 macro-tiles over 512x512), direct-global WMMA
  k_gemm_dist<<<(Bc * 256 * 32) / 256, 256, 0, stream>>>(dn, acc);
  // pairwise projection of feature points
  k_project<<<(Bc * Nc + 255) / 256, 256, 0, stream>>>(points, depths, poses, Km,
                                                        proj_p, invis_p);
  // corner pipeline
  k_gray<<<(HWB + 255) / 256, 256, 0, stream>>>(imgs, gray);
  k_grad<<<(HWB + 255) / 256, 256, 0, stream>>>(gray, gxb, gyb);
  k_resp<<<(HWB + 255) / 256, 256, 0, stream>>>(gxb, gyb, resp);
  k_nms<<<(HWB + 255) / 256, 256, 0, stream>>>(resp, nmsv);
  dim3 gcb(Wc / 8, Hc / 8, Bc);
  k_blockcand<<<gcb, 64, 0, stream>>>(nmsv, cand_key, cand_cnt);
  k_topk<<<Bc, 512, 0, stream>>>(cand_key, cvals, cpts);
  k_project<<<(Bc * Nc + 255) / 256, 256, 0, stream>>>(cpts, depths, poses, Km,
                                                        proj_c, invis_c);
  k_scatter<<<(Bc * Bc * Nc + 255) / 256, 256, 0, stream>>>(proj_c, invis_c, cvals, target);
  k_bce<<<(HWB + 255) / 256, 256, 0, stream>>>(target, pointness, acc);
  // match: 64 pairs x 64 regions, TDM-staged LDS operands + WMMA
  k_gemm_match<<<Bc * Bc * 64, 128, 0, stream>>>(dn, points, proj_p, invis_p, acc);
  k_final<<<1, 1, 0, stream>>>(acc, out);
}
